// RotaryAttention_28054726377953
// MI455X (gfx1250) — matrix operand order repaired and checked by reading, not compiled, not run on hardware
//
#include <hip/hip_runtime.h>
#include <hip/hip_bf16.h>

#define D_MODEL  2048
#define NHEADS   16
#define HEAD_DIM 128
#define BATCH    2
#define SEQLEN   2048
#define MTOT     (BATCH * SEQLEN)   // 4096
#define NQKV     (3 * D_MODEL)      // 6144

typedef __attribute__((ext_vector_type(16))) __bf16 v16bf;
typedef __attribute__((ext_vector_type(8)))  __bf16 v8bf;
typedef __attribute__((ext_vector_type(4)))  __bf16 v4bf;
typedef __attribute__((ext_vector_type(8)))  float  v8f;
typedef __attribute__((ext_vector_type(4)))  float  v4f;
typedef __attribute__((ext_vector_type(4)))  int    v4i;

union BF16Frag { v16bf v; v8bf h[2]; };

static __device__ inline v8f wmma_bf16(const BF16Frag& a, const BF16Frag& b, v8f c) {
  // (neg_a, A, neg_b, B, c_mod, C, reuse_a, reuse_b)
  return __builtin_amdgcn_wmma_f32_16x16x32_bf16(false, a.v, false, b.v, (short)0, c,
                                                 false, false);
}

// --- gfx1250 async global->LDS copy (ASYNCcnt path), with safe fallback -----
#if __has_builtin(__builtin_amdgcn_global_load_async_to_lds_b128)
#define HAVE_ASYNC_LDS 1
static __device__ inline void lds_copy16(void* lds_dst, const void* gsrc) {
  __builtin_amdgcn_global_load_async_to_lds_b128(
      (__attribute__((address_space(1))) v4i*)(gsrc),
      (__attribute__((address_space(3))) v4i*)(lds_dst), 0, 0);
}
#else
#define HAVE_ASYNC_LDS 0
static __device__ inline void lds_copy16(void* lds_dst, const void* gsrc) {
  *(v8bf*)lds_dst = *(const v8bf*)gsrc;
}
#endif

static __device__ inline void lds_copy_wait() {
#if HAVE_ASYNC_LDS
#if __has_builtin(__builtin_amdgcn_s_wait_asynccnt)
  __builtin_amdgcn_s_wait_asynccnt(0);
#else
  asm volatile("s_wait_asynccnt 0x0" ::: "memory");
#endif
#endif
}

// ---------------------------------------------------------------------------
// Kernel 1: QKV projection. C[m,n] = sum_k X[m,k] * W[n,k] + bias[n]
// Epilogue scatters bf16 into Q [b,h,s,d], K [b,h,s,d], V^T [b,h,d,s].
// ---------------------------------------------------------------------------
__global__ __launch_bounds__(256) void qkv_gemm_kernel(
    const float* __restrict__ X, const float* __restrict__ W,
    const float* __restrict__ bias,
    __bf16* __restrict__ Qg, __bf16* __restrict__ Kg, __bf16* __restrict__ VTg)
{
  __shared__ __bf16 As[128][40];
  __shared__ __bf16 Bs[128][40];
  const int tid  = threadIdx.x;
  const int wave = tid >> 5, lane = tid & 31;
  const int kh   = lane >> 4, ln = lane & 15;
  const int Mblk = blockIdx.y * 128;
  const int Nblk = blockIdx.x * 128;
  const int moff = (wave >> 1) * 32;
  const int noff = (wave & 1) * 64;

  v8f acc[2][4];
  for (int mt = 0; mt < 2; ++mt)
    for (int nt = 0; nt < 4; ++nt)
      for (int i = 0; i < 8; ++i) acc[mt][nt][i] = 0.0f;

  for (int k0 = 0; k0 < D_MODEL; k0 += 32) {
    // Stage A and B tiles (fp32 -> bf16): 128 rows x 32 cols each.
    {
      const int r0 = tid >> 3, c = (tid & 7) * 4;
      for (int it = 0; it < 4; ++it) {
        const int r = r0 + it * 32;
        const float* pa = X + (size_t)(Mblk + r) * D_MODEL + k0 + c;
        const float* pb = W + (size_t)(Nblk + r) * D_MODEL + k0 + c;
        if (k0 + 32 < D_MODEL) {          // prefetch next k-panel
          __builtin_prefetch(pa + 32, 0, 1);
          __builtin_prefetch(pb + 32, 0, 1);
        }
        v4f fa = *(const v4f*)pa;
        v4f fb = *(const v4f*)pb;
        v4bf ba, bb;
        for (int j = 0; j < 4; ++j) { ba[j] = (__bf16)fa[j]; bb[j] = (__bf16)fb[j]; }
        *(v4bf*)&As[r][c] = ba;
        *(v4bf*)&Bs[r][c] = bb;
      }
    }
    __syncthreads();

    BF16Frag a[2], b[4];
    for (int mt = 0; mt < 2; ++mt) {
      const int r = moff + mt * 16 + ln;
      a[mt].h[0] = *(const v8bf*)&As[r][kh * 8];
      a[mt].h[1] = *(const v8bf*)&As[r][16 + kh * 8];
    }
    for (int nt = 0; nt < 4; ++nt) {
      const int r = noff + nt * 16 + ln;
      b[nt].h[0] = *(const v8bf*)&Bs[r][kh * 8];
      b[nt].h[1] = *(const v8bf*)&Bs[r][16 + kh * 8];
    }
    for (int mt = 0; mt < 2; ++mt)
      for (int nt = 0; nt < 4; ++nt)
        acc[mt][nt] = wmma_bf16(a[mt], b[nt], acc[mt][nt]);
    __syncthreads();
  }

  for (int mt = 0; mt < 2; ++mt) {
    for (int nt = 0; nt < 4; ++nt) {
      const int n   = Nblk + noff + nt * 16 + ln;
      const float bv = bias[n];
      const int sec = n >> 11;          // 0=q 1=k 2=v
      const int hh  = (n >> 7) & 15;    // head
      const int d   = n & 127;          // within head
      for (int i = 0; i < 8; ++i) {
        const int m    = Mblk + moff + mt * 16 + i + 8 * kh;
        const int bidx = m >> 11, sidx = m & (SEQLEN - 1);
        const __bf16 val = (__bf16)(acc[mt][nt][i] + bv);
        if (sec == 0)
          Qg[(((size_t)bidx * NHEADS + hh) * SEQLEN + sidx) * HEAD_DIM + d] = val;
        else if (sec == 1)
          Kg[(((size_t)bidx * NHEADS + hh) * SEQLEN + sidx) * HEAD_DIM + d] = val;
        else
          VTg[(((size_t)bidx * NHEADS + hh) * HEAD_DIM + d) * SEQLEN + sidx] = val;
      }
    }
  }
}

// ---------------------------------------------------------------------------
// Kernel 2: in-place rotary on [B*H, S, 128]; pairs (d, d+64). outscale lets
// us fold the 1/sqrt(HEAD_DIM) score scaling into Q.
// ---------------------------------------------------------------------------
__global__ __launch_bounds__(256) void rotary_kernel(__bf16* __restrict__ T,
                                                     float outscale)
{
  const int idx = blockIdx.x * 256 + threadIdx.x;   // over B*H*S*64
  const int p   = idx & 63;
  const int s   = (idx >> 6) & (SEQLEN - 1);
  const int bh  = idx >> 17;
  const size_t base = ((size_t)bh * SEQLEN + s) * HEAD_DIM;
  // inv_freq = 10000^{-(2p)/128};  ln(10000) = 9.210340371976184
  const float inv = __expf(-(float)(2 * p) * (9.210340372f / 128.0f));
  const float th  = (float)s * inv;
  float sn, cs;
  __sincosf(th, &sn, &cs);
  const float x1 = (float)T[base + p];
  const float x2 = (float)T[base + 64 + p];
  T[base + p]      = (__bf16)((x1 * cs - x2 * sn) * outscale);
  T[base + 64 + p] = (__bf16)((x2 * cs + x1 * sn) * outscale);
}

// ---------------------------------------------------------------------------
// Kernel 3: causal flash attention. Block = 128 query rows, 8 waves x 16 rows.
// Q fragments in registers; K and V^T tiles staged via async global->LDS per
// 32-key step; P transposed C-layout -> A-layout through per-wave LDS.
// ---------------------------------------------------------------------------
__global__ __launch_bounds__(256) void attn_kernel(
    const __bf16* __restrict__ Qg, const __bf16* __restrict__ Kg,
    const __bf16* __restrict__ VTg, __bf16* __restrict__ Ctx)
{
  __shared__ __bf16 Ks[32][136];     // [key][d]
  __shared__ __bf16 VTs[128][40];    // [d][key]
  __shared__ __bf16 Ps[8][16][40];   // per-wave P tile [row][key]

  const int tid  = threadIdx.x;
  const int wave = tid >> 5, lane = tid & 31;
  const int kh   = lane >> 4, ln = lane & 15;
  const int qblk = blockIdx.x * 128;
  const int bh   = blockIdx.y;
  const int b    = bh >> 4, hh = bh & 15;

  // Q A-fragments: lane owns query row (qblk + wave*16 + ln), 4 k-steps of 32.
  BF16Frag aq[4];
  {
    const __bf16* qp =
        Qg + ((size_t)bh * SEQLEN + qblk + wave * 16 + ln) * HEAD_DIM;
    for (int dsi = 0; dsi < 4; ++dsi) {
      aq[dsi].h[0] = *(const v8bf*)(qp + dsi * 32 + kh * 8);
      aq[dsi].h[1] = *(const v8bf*)(qp + dsi * 32 + 16 + kh * 8);
    }
  }

  v8f O[8];
  for (int j = 0; j < 8; ++j)
    for (int i = 0; i < 8; ++i) O[j][i] = 0.0f;
  float mrun[8], lrun[8];
  for (int i = 0; i < 8; ++i) { mrun[i] = -1e30f; lrun[i] = 0.0f; }

  const int nsteps = (qblk + 128) / 32;
  for (int kb = 0; kb < nsteps; ++kb) {
    const int kt0 = kb * 32;
    // Stage K (32x128) and V^T (128x32) through the async-to-LDS path.
    for (int it = 0; it < 2; ++it) {
      const int idx = it * 256 + tid;
      const int r = idx >> 4, c = (idx & 15) * 8;
      lds_copy16(&Ks[r][c],
                 Kg + ((size_t)bh * SEQLEN + kt0 + r) * HEAD_DIM + c);
      const int r2 = idx >> 2, c2 = (idx & 3) * 8;
      lds_copy16(&VTs[r2][c2],
                 VTg + ((size_t)bh * HEAD_DIM + r2) * SEQLEN + kt0 + c2);
    }
    lds_copy_wait();
    __syncthreads();

    // Scores: two 16x16 tiles over this 32-key block.
    v8f t0, t1;
    for (int i = 0; i < 8; ++i) { t0[i] = 0.0f; t1[i] = 0.0f; }
    for (int dsi = 0; dsi < 4; ++dsi) {
      BF16Frag b0, b1;
      b0.h[0] = *(const v8bf*)&Ks[ln][dsi * 32 + kh * 8];
      b0.h[1] = *(const v8bf*)&Ks[ln][dsi * 32 + 16 + kh * 8];
      b1.h[0] = *(const v8bf*)&Ks[16 + ln][dsi * 32 + kh * 8];
      b1.h[1] = *(const v8bf*)&Ks[16 + ln][dsi * 32 + 16 + kh * 8];
      t0 = wmma_bf16(aq[dsi], b0, t0);
      t1 = wmma_bf16(aq[dsi], b1, t1);
    }

    // Online softmax; C layout: VGPR i -> row (i + 8*kh), lane -> column.
    float alpha[8];
    for (int i = 0; i < 8; ++i) {
      const int qr = qblk + wave * 16 + i + 8 * kh;
      float s0 = (kt0 + ln      <= qr) ? t0[i] : -1e30f;
      float s1 = (kt0 + 16 + ln <= qr) ? t1[i] : -1e30f;
      float loc = fmaxf(s0, s1);
      for (int off = 8; off > 0; off >>= 1)
        loc = fmaxf(loc, __shfl_xor(loc, off, 16));
      const float mn = fmaxf(mrun[i], loc);
      const float al = __expf(mrun[i] - mn);
      const float p0 = __expf(s0 - mn);
      const float p1 = __expf(s1 - mn);
      float rs = p0 + p1;
      for (int off = 8; off > 0; off >>= 1) rs += __shfl_xor(rs, off, 16);
      lrun[i] = lrun[i] * al + rs;
      mrun[i] = mn;
      alpha[i] = al;
      Ps[wave][i + 8 * kh][ln]      = (__bf16)p0;
      Ps[wave][i + 8 * kh][ln + 16] = (__bf16)p1;
    }
    for (int j = 0; j < 8; ++j)
      for (int i = 0; i < 8; ++i) O[j][i] *= alpha[i];

    // Intra-wave LDS transpose hand-off: wait for our ds stores to land.
    asm volatile("s_wait_dscnt 0x0" ::: "memory");
    BF16Frag ap;
    ap.h[0] = *(const v8bf*)&Ps[wave][ln][kh * 8];
    ap.h[1] = *(const v8bf*)&Ps[wave][ln][16 + kh * 8];

    for (int dt = 0; dt < 8; ++dt) {
      BF16Frag bv;
      bv.h[0] = *(const v8bf*)&VTs[dt * 16 + ln][kh * 8];
      bv.h[1] = *(const v8bf*)&VTs[dt * 16 + ln][16 + kh * 8];
      O[dt] = wmma_bf16(ap, bv, O[dt]);
    }
    __syncthreads();
  }

  // Normalize and emit ctx as bf16 [b, s, h*128 + d] = [4096, 2048].
  for (int dt = 0; dt < 8; ++dt) {
    for (int i = 0; i < 8; ++i) {
      const int m   = b * SEQLEN + qblk + wave * 16 + i + 8 * kh;
      const int col = hh * HEAD_DIM + dt * 16 + ln;
      Ctx[(size_t)m * D_MODEL + col] = (__bf16)(O[dt][i] / lrun[i]);
    }
  }
}

// ---------------------------------------------------------------------------
// Kernel 4: output projection. Out[m,n] = sum_k Ctx[m,k] * W[n,k] + bias[n]
// A tile (bf16) staged via async global->LDS; B tile converted fp32->bf16.
// ---------------------------------------------------------------------------
__global__ __launch_bounds__(256) void out_gemm_kernel(
    const __bf16* __restrict__ Ctx, const float* __restrict__ W,
    const float* __restrict__ bias, float* __restrict__ Out)
{
  __shared__ __bf16 As[128][40];
  __shared__ __bf16 Bs[128][40];
  const int tid  = threadIdx.x;
  const int wave = tid >> 5, lane = tid & 31;
  const int kh   = lane >> 4, ln = lane & 15;
  const int Mblk = blockIdx.y * 128;
  const int Nblk = blockIdx.x * 128;
  const int moff = (wave >> 1) * 32;
  const int noff = (wave & 1) * 64;

  v8f acc[2][4];
  for (int mt = 0; mt < 2; ++mt)
    for (int nt = 0; nt < 4; ++nt)
      for (int i = 0; i < 8; ++i) acc[mt][nt][i] = 0.0f;

  for (int k0 = 0; k0 < D_MODEL; k0 += 32) {
    // Stage A (bf16, async path) and B (fp32 -> bf16 via VGPRs).
    for (int it = 0; it < 2; ++it) {
      const int idx = it * 256 + tid;
      const int r = idx >> 2, c = (idx & 3) * 8;
      lds_copy16(&As[r][c], Ctx + (size_t)(Mblk + r) * D_MODEL + k0 + c);
    }
    {
      const int r0 = tid >> 3, c = (tid & 7) * 4;
      for (int it = 0; it < 4; ++it) {
        const int r = r0 + it * 32;
        const float* pb = W + (size_t)(Nblk + r) * D_MODEL + k0 + c;
        if (k0 + 32 < D_MODEL) __builtin_prefetch(pb + 32, 0, 1);
        v4f g = *(const v4f*)pb;
        v4bf bb;
        for (int j = 0; j < 4; ++j) bb[j] = (__bf16)g[j];
        *(v4bf*)&Bs[r][c] = bb;
      }
    }
    lds_copy_wait();
    __syncthreads();

    BF16Frag a[2], b[4];
    for (int mt = 0; mt < 2; ++mt) {
      const int r = moff + mt * 16 + ln;
      a[mt].h[0] = *(const v8bf*)&As[r][kh * 8];
      a[mt].h[1] = *(const v8bf*)&As[r][16 + kh * 8];
    }
    for (int nt = 0; nt < 4; ++nt) {
      const int r = noff + nt * 16 + ln;
      b[nt].h[0] = *(const v8bf*)&Bs[r][kh * 8];
      b[nt].h[1] = *(const v8bf*)&Bs[r][16 + kh * 8];
    }
    for (int mt = 0; mt < 2; ++mt)
      for (int nt = 0; nt < 4; ++nt)
        acc[mt][nt] = wmma_bf16(a[mt], b[nt], acc[mt][nt]);
    __syncthreads();
  }

  for (int mt = 0; mt < 2; ++mt) {
    for (int nt = 0; nt < 4; ++nt) {
      const int n = Nblk + noff + nt * 16 + ln;
      const float bv = bias[n];
      for (int i = 0; i < 8; ++i) {
        const int m = Mblk + moff + mt * 16 + i + 8 * kh;
        Out[(size_t)m * D_MODEL + n] = acc[mt][nt][i] + bv;
      }
    }
  }
}

// ---------------------------------------------------------------------------
extern "C" void kernel_launch(void* const* d_in, const int* in_sizes, int n_in,
                              void* d_out, int out_size, void* d_ws, size_t ws_size,
                              hipStream_t stream) {
  (void)in_sizes; (void)n_in; (void)out_size; (void)ws_size;
  const float* x      = (const float*)d_in[0];
  // d_in[1] = attn_mask (causal; computed analytically in-kernel)
  const float* Wqkv_w = (const float*)d_in[2];
  const float* Wqkv_b = (const float*)d_in[3];
  const float* out_w  = (const float*)d_in[4];
  const float* out_b  = (const float*)d_in[5];
  float* out = (float*)d_out;

  char* ws = (char*)d_ws;
  const size_t qbytes =
      (size_t)BATCH * NHEADS * SEQLEN * HEAD_DIM * sizeof(__bf16);  // 16 MB
  __bf16* Qg  = (__bf16*)(ws);
  __bf16* Kg  = (__bf16*)(ws + qbytes);
  __bf16* VTg = (__bf16*)(ws + 2 * qbytes);
  __bf16* Ctx = (__bf16*)(ws + 3 * qbytes);

  qkv_gemm_kernel<<<dim3(NQKV / 128, MTOT / 128), 256, 0, stream>>>(
      x, Wqkv_w, Wqkv_b, Qg, Kg, VTg);

  const int rthreads = BATCH * NHEADS * SEQLEN * 64;
  rotary_kernel<<<rthreads / 256, 256, 0, stream>>>(Qg, 0.0883883476483184f);
  rotary_kernel<<<rthreads / 256, 256, 0, stream>>>(Kg, 1.0f);

  attn_kernel<<<dim3(SEQLEN / 128, BATCH * NHEADS), 256, 0, stream>>>(
      Qg, Kg, VTg, Ctx);

  out_gemm_kernel<<<dim3(D_MODEL / 128, MTOT / 128), 256, 0, stream>>>(
      Ctx, out_w, out_b, out);
}
